// mm_tab_78331613544645
// MI455X (gfx1250) — compile-verified, matrix operand order checked by reading, not run on hardware
//
#include <hip/hip_runtime.h>
#include <hip/hip_bf16.h>
#include <math.h>

// ---------------------------------------------------------------------------
// TabTransformer fused kernel for gfx1250 (MI455X).
//   B=8192, N=7 tokens, DIM=128, HEADS=8, DH=16, DEPTH=6, HID=512.
// Strategy: one workgroup (8 wave32) owns 8 batch items (M = 64 rows, token
// slot 7 zero padded). Activations live in LDS for all 6 layers; all GEMMs run
// on v_wmma_f32_16x16x32_bf16 with fp32 accumulation; weights are pre-packed
// to bf16 WMMA B-fragment layout in d_ws (L2-resident, ~3MB).
// ---------------------------------------------------------------------------

typedef __bf16 bf16_t;
typedef bf16_t v16bf __attribute__((ext_vector_type(16)));
typedef bf16_t v8bf  __attribute__((ext_vector_type(8)));
typedef float  v8f   __attribute__((ext_vector_type(8)));

#define DEPTH_L   6
#define DIMM      128
#define HIDD      512
#define NTOK      7
#define MROWS     64          // 8 batch items * 8 token slots
#define BATCH_WG  8
#define BATCH_N   8192

// packed weight element counts (bf16) per layer
#define QKV_E (128*384)
#define OUT_E (128*128)
#define FF1_E (128*1024)
#define FF2_E (512*128)
#define LAYER_E (QKV_E + OUT_E + FF1_E + FF2_E)   // 262144 elems = 512KB bf16

// ---------------------------------------------------------------------------
// WMMA helpers
// ---------------------------------------------------------------------------
__device__ __forceinline__ v8f wmma_bf16(v16bf a, v16bf b, v8f c) {
  // (neg_a, A, neg_b, B, c_mod, C, reuse_a, reuse_b)
  return __builtin_amdgcn_wmma_f32_16x16x32_bf16(false, a, false, b,
                                                 (short)0, c, false, false);
}

// A fragment (16x32 bf16) from row-major LDS activation tile.
// ISA 16-bit A layout: lane group g = lane>>4; halves 0..7 -> K = g*8+h,
// halves 8..15 -> K = 16 + g*8 + (h-8)  => two contiguous 16B reads per lane.
__device__ __forceinline__ v16bf lds_a_frag(const bf16_t* base, int ld,
                                            int mt, int kt, int lane) {
  const int row = mt * 16 + (lane & 15);
  const int g   = lane >> 4;
  const bf16_t* p = base + row * ld + kt * 32 + g * 8;
  v8bf lo = *(const v8bf*)p;
  v8bf hi = *(const v8bf*)(p + 16);
  v16bf r;
#pragma unroll
  for (int i = 0; i < 8; ++i) { r[i] = lo[i]; r[i + 8] = hi[i]; }
  return r;
}

// B fragment (32x16 bf16) from pre-packed global weights: one 32B load/lane.
__device__ __forceinline__ v16bf g_b_frag(const bf16_t* packed, int ktiles,
                                          int nt, int kt, int lane) {
  const bf16_t* p = packed + ((long)(nt * ktiles + kt) * 32 + lane) * 16;
  return *(const v16bf*)p;
}

// C/D layout: lane 0-15 -> N=lane, M=v; lane 16-31 -> N=lane-16, M=8+v.
__device__ __forceinline__ void store_tile_bf16(bf16_t* dst, int ld, int mt,
                                                int ntcol, v8f acc, int lane) {
  const int col   = ntcol * 16 + (lane & 15);
  const int rbase = mt * 16 + ((lane >> 4) * 8);
#pragma unroll
  for (int v = 0; v < 8; ++v) dst[(rbase + v) * ld + col] = (bf16_t)acc[v];
}

__device__ __forceinline__ float gelu_exact(float x) {
  return 0.5f * x * (1.0f + erff(x * 0.70710678118654752f));
}

// ---------------------------------------------------------------------------
// Weight pre-pack: fp32 row-major [K,N] -> bf16 WMMA B-fragment tiles.
// packed[(nt*ktiles + kt)*512 + lane*16 + h] = W[kt*32 + kpat(lane,h)][nt*16 + (lane&15)]
// kpat(lane,h) = 16*(h>>3) + 8*(lane>>4) + (h&7)
// ---------------------------------------------------------------------------
__global__ void pack_weights(const float* __restrict__ w_qkv,
                             const float* __restrict__ w_out,
                             const float* __restrict__ w_ff1,
                             const float* __restrict__ w_ff2,
                             bf16_t* __restrict__ packed) {
  long gid = (long)blockIdx.x * blockDim.x + threadIdx.x;
  const long total = (long)DEPTH_L * LAYER_E;
  if (gid >= total) return;
  const int layer = (int)(gid / LAYER_E);
  int e = (int)(gid % LAYER_E);
  const float* src; int K, Nw, rel;
  if (e < QKV_E)                    { src = w_qkv + (long)layer*QKV_E; K=128; Nw=384;  rel = e; }
  else if (e < QKV_E+OUT_E)         { src = w_out + (long)layer*OUT_E; K=128; Nw=128;  rel = e - QKV_E; }
  else if (e < QKV_E+OUT_E+FF1_E)   { src = w_ff1 + (long)layer*FF1_E; K=128; Nw=1024; rel = e - (QKV_E+OUT_E); }
  else                              { src = w_ff2 + (long)layer*FF2_E; K=512; Nw=128;  rel = e - (QKV_E+OUT_E+FF1_E); }
  const int frag = rel >> 9;
  const int lane = (rel >> 4) & 31;
  const int h    = rel & 15;
  const int ktiles = K >> 5;
  const int nt = frag / ktiles;
  const int kt = frag - nt * ktiles;
  const int n  = nt * 16 + (lane & 15);
  const int k  = kt * 32 + ((h >> 3) << 4) + ((lane >> 4) << 3) + (h & 7);
  packed[gid] = (bf16_t)src[(long)k * Nw + n];
}

// ---------------------------------------------------------------------------
// Fused 6-layer transformer. 256 threads = 8 waves. Grid = B/8 workgroups.
// ---------------------------------------------------------------------------
__global__ void __launch_bounds__(256)
tab_transformer_fused(const int* __restrict__ ids,
                      const float* __restrict__ embeds,
                      const float* __restrict__ ln1_g, const float* __restrict__ ln1_b,
                      const float* __restrict__ b_out,
                      const float* __restrict__ ln2_g, const float* __restrict__ ln2_b,
                      const float* __restrict__ b_ff1, const float* __restrict__ b_ff2,
                      const bf16_t* __restrict__ packed,
                      float* __restrict__ out) {
  extern __shared__ unsigned char smem[];
  float*  Xs   = (float*)smem;                                        // 64x128 f32  (32KB)
  bf16_t* Hs   = (bf16_t*)(smem + MROWS*DIMM*4);                      // 64x128 bf16 (16KB)
  bf16_t* QKVs = (bf16_t*)(smem + MROWS*DIMM*4 + MROWS*DIMM*2);       // 64x384 bf16 (48KB)
  bf16_t* Gs   = (bf16_t*)(smem + MROWS*DIMM*4 + MROWS*DIMM*2 + MROWS*384*2); // 64x128 (16KB)

  const int t    = threadIdx.x;
  const int lane = t & 31;
  const int w    = t >> 5;
  const int wg   = blockIdx.x;

  // ---- embedding gather: X[row][col], token slot 7 zero-padded ----
  for (int idx = t; idx < MROWS * DIMM; idx += 256) {
    const int row = idx >> 7, col = idx & 127;
    const int tok = row & 7;
    float v = 0.f;
    if (tok < NTOK) {
      const int b  = wg * BATCH_WG + (row >> 3);
      const int id = ids[b * NTOK + tok];
      v = embeds[(long)id * DIMM + col];
    }
    Xs[idx] = v;
  }
  __syncthreads();

  for (int layer = 0; layer < DEPTH_L; ++layer) {
    const bf16_t* Wqkv = packed + (long)layer * LAYER_E;
    const bf16_t* Wout = Wqkv + QKV_E;
    const bf16_t* Wff1 = Wout + OUT_E;
    const bf16_t* Wff2 = Wff1 + FF1_E;

    // warm next layer's weights into L2 (global_prefetch_b8)
    if (layer + 1 < DEPTH_L && t < 64)
      __builtin_prefetch(packed + (long)(layer + 1) * LAYER_E + (long)t * 4096, 0, 1);

    // ---- LayerNorm 1 -> Hs (bf16) ----
    if (t < MROWS) {
      const float* xr = Xs + t * DIMM;
      const float* gg = ln1_g + layer * DIMM;
      const float* bb = ln1_b + layer * DIMM;
      float mu = 0.f;
      for (int d = 0; d < DIMM; ++d) mu += xr[d];
      mu *= (1.f / DIMM);
      float var = 0.f;
      for (int d = 0; d < DIMM; ++d) { float dd = xr[d] - mu; var += dd * dd; }
      var *= (1.f / DIMM);
      const float inv = rsqrtf(var + 1e-5f);
      for (int d = 0; d < DIMM; ++d)
        Hs[t * DIMM + d] = (bf16_t)((xr[d] - mu) * inv * gg[d] + bb[d]);
    }
    __syncthreads();

    // ---- QKV: Hs[64x128] @ Wqkv[128x384] -> QKVs (96 tiles, 12 per wave) ----
    for (int i = 0; i < 12; ++i) {
      const int tile = w + 8 * i;
      const int mt = tile / 24, nt = tile % 24;
      v8f acc = {0.f,0.f,0.f,0.f,0.f,0.f,0.f,0.f};
#pragma unroll
      for (int kt = 0; kt < 4; ++kt) {
        v16bf a = lds_a_frag(Hs, DIMM, mt, kt, lane);
        v16bf b = g_b_frag(Wqkv, 4, nt, kt, lane);
        acc = wmma_bf16(a, b, acc);
      }
      store_tile_bf16(QKVs, 384, mt, nt, acc, lane);
    }
    __syncthreads();

    // ---- attention (7x7 per head, fp32 VALU) -> Hs ----
    if (t < 64) {
      const int bl = t >> 3, h = t & 7;
      const int r0 = bl * 8;
      float s[7][7];
      for (int i = 0; i < 7; ++i)
        for (int j = 0; j < 7; ++j) {
          float acc = 0.f;
          for (int d = 0; d < 16; ++d) {
            const float qv = (float)QKVs[(r0 + i) * 384 + h * 16 + d];
            const float kv = (float)QKVs[(r0 + j) * 384 + 128 + h * 16 + d];
            acc += qv * kv;
          }
          s[i][j] = acc * 0.25f;   // DH^-0.5
        }
      for (int i = 0; i < 7; ++i) {
        float m = s[i][0];
        for (int j = 1; j < 7; ++j) m = fmaxf(m, s[i][j]);
        float den = 0.f;
        for (int j = 0; j < 7; ++j) { s[i][j] = __expf(s[i][j] - m); den += s[i][j]; }
        const float inv = 1.f / den;
        for (int d = 0; d < 16; ++d) {
          float o = 0.f;
          for (int j = 0; j < 7; ++j)
            o += s[i][j] * (float)QKVs[(r0 + j) * 384 + 256 + h * 16 + d];
          Hs[(r0 + i) * DIMM + h * 16 + d] = (bf16_t)(o * inv);
        }
      }
      for (int d = 0; d < 16; ++d)   // padded token row -> 0
        Hs[(r0 + 7) * DIMM + h * 16 + d] = (bf16_t)0.f;
    }
    __syncthreads();

    // ---- out-proj + bias + residual into Xs (32 tiles, 4 per wave) ----
    for (int i = 0; i < 4; ++i) {
      const int tile = w + 8 * i;
      const int mt = tile >> 3, nt = tile & 7;
      v8f acc = {0.f,0.f,0.f,0.f,0.f,0.f,0.f,0.f};
#pragma unroll
      for (int kt = 0; kt < 4; ++kt) {
        v16bf a = lds_a_frag(Hs, DIMM, mt, kt, lane);
        v16bf b = g_b_frag(Wout, 4, nt, kt, lane);
        acc = wmma_bf16(a, b, acc);
      }
      const int col   = nt * 16 + (lane & 15);
      const int rbase = mt * 16 + ((lane >> 4) * 8);
      const float bias = b_out[layer * DIMM + col];
#pragma unroll
      for (int v = 0; v < 8; ++v)
        Xs[(rbase + v) * DIMM + col] += acc[v] + bias;
    }
    __syncthreads();

    // ---- LayerNorm 2 -> Hs ----
    if (t < MROWS) {
      const float* xr = Xs + t * DIMM;
      const float* gg = ln2_g + layer * DIMM;
      const float* bb = ln2_b + layer * DIMM;
      float mu = 0.f;
      for (int d = 0; d < DIMM; ++d) mu += xr[d];
      mu *= (1.f / DIMM);
      float var = 0.f;
      for (int d = 0; d < DIMM; ++d) { float dd = xr[d] - mu; var += dd * dd; }
      var *= (1.f / DIMM);
      const float inv = rsqrtf(var + 1e-5f);
      for (int d = 0; d < DIMM; ++d)
        Hs[t * DIMM + d] = (bf16_t)((xr[d] - mu) * inv * gg[d] + bb[d]);
    }
    __syncthreads();

    // ---- FFN (GEGLU), hidden processed in 4 chunks of 128 ----
    v8f ytile[4];
#pragma unroll
    for (int i = 0; i < 4; ++i) ytile[i] = (v8f){0.f,0.f,0.f,0.f,0.f,0.f,0.f,0.f};

    for (int c = 0; c < 4; ++c) {
      // a/g tiles for this chunk: u = Hs @ Wff1 (+bias); Gs = a * gelu(g)
      for (int i = 0; i < 4; ++i) {
        const int tile = w + 8 * i;
        const int mt = tile >> 3, nt = tile & 7;
        v8f accA = {0.f,0.f,0.f,0.f,0.f,0.f,0.f,0.f};
        v8f accG = {0.f,0.f,0.f,0.f,0.f,0.f,0.f,0.f};
#pragma unroll
        for (int kt = 0; kt < 4; ++kt) {
          v16bf a  = lds_a_frag(Hs, DIMM, mt, kt, lane);
          v16bf ba = g_b_frag(Wff1, 4, c * 8 + nt, kt, lane);        // a-half cols
          v16bf bg = g_b_frag(Wff1, 4, 32 + c * 8 + nt, kt, lane);   // g-half cols
          accA = wmma_bf16(a, ba, accA);
          accG = wmma_bf16(a, bg, accG);
        }
        const int col   = nt * 16 + (lane & 15);
        const int rbase = mt * 16 + ((lane >> 4) * 8);
        const float biasA = b_ff1[layer * 2 * HIDD + c * 128 + col];
        const float biasG = b_ff1[layer * 2 * HIDD + HIDD + c * 128 + col];
#pragma unroll
        for (int v = 0; v < 8; ++v) {
          const float av = accA[v] + biasA;
          const float gv = accG[v] + biasG;
          Gs[(rbase + v) * DIMM + col] = (bf16_t)(av * gelu_exact(gv));
        }
      }
      __syncthreads();

      // ff2 partial: y += Gs[64x128] @ Wff2[chunk rows, 128]
      for (int i = 0; i < 4; ++i) {
        const int tile = w + 8 * i;
        const int mt = tile >> 3, nt = tile & 7;
        v8f acc = ytile[i];
#pragma unroll
        for (int kk = 0; kk < 4; ++kk) {
          v16bf a = lds_a_frag(Gs, DIMM, mt, kk, lane);
          v16bf b = g_b_frag(Wff2, 16, nt, c * 4 + kk, lane);
          acc = wmma_bf16(a, b, acc);
        }
        ytile[i] = acc;
      }
      __syncthreads();   // Gs reuse by next chunk
    }

    // ---- ff2 bias + residual into Xs ----
    for (int i = 0; i < 4; ++i) {
      const int tile = w + 8 * i;
      const int mt = tile >> 3, nt = tile & 7;
      const int col   = nt * 16 + (lane & 15);
      const int rbase = mt * 16 + ((lane >> 4) * 8);
      const float bias = b_ff2[layer * DIMM + col];
#pragma unroll
      for (int v = 0; v < 8; ++v)
        Xs[(rbase + v) * DIMM + col] += ytile[i][v] + bias;
    }
    __syncthreads();
  }

  // ---- write output (fp32, only real tokens) ----
  for (int idx = t; idx < MROWS * DIMM; idx += 256) {
    const int row = idx >> 7, col = idx & 127;
    const int tok = row & 7;
    if (tok < NTOK) {
      const int b = wg * BATCH_WG + (row >> 3);
      out[((long)b * NTOK + tok) * DIMM + col] = Xs[idx];
    }
  }
}

// ---------------------------------------------------------------------------
extern "C" void kernel_launch(void* const* d_in, const int* in_sizes, int n_in,
                              void* d_out, int out_size, void* d_ws, size_t ws_size,
                              hipStream_t stream) {
  const int*   ids    = (const int*)  d_in[0];
  const float* embeds = (const float*)d_in[1];
  const float* ln1_g  = (const float*)d_in[2];
  const float* ln1_b  = (const float*)d_in[3];
  const float* w_qkv  = (const float*)d_in[4];
  const float* w_out  = (const float*)d_in[5];
  const float* b_out  = (const float*)d_in[6];
  const float* ln2_g  = (const float*)d_in[7];
  const float* ln2_b  = (const float*)d_in[8];
  const float* w_ff1  = (const float*)d_in[9];
  const float* b_ff1  = (const float*)d_in[10];
  const float* w_ff2  = (const float*)d_in[11];
  const float* b_ff2  = (const float*)d_in[12];

  bf16_t* packed = (bf16_t*)d_ws;   // 6 * 262144 bf16 = 3 MB

  const long total = (long)DEPTH_L * LAYER_E;
  pack_weights<<<(int)((total + 255) / 256), 256, 0, stream>>>(
      w_qkv, w_out, w_ff1, w_ff2, packed);

  const size_t lds_bytes = (size_t)MROWS * DIMM * 4      // Xs
                         + (size_t)MROWS * DIMM * 2      // Hs
                         + (size_t)MROWS * 384  * 2      // QKVs
                         + (size_t)MROWS * DIMM * 2;     // Gs  = 114688 B
  tab_transformer_fused<<<BATCH_N / BATCH_WG, 256, lds_bytes, stream>>>(
      ids, embeds, ln1_g, ln1_b, b_out, ln2_g, ln2_b, b_ff1, b_ff2,
      packed, (float*)d_out);
}
